// PrototypeObservationExplainer_24318104830570
// MI455X (gfx1250) — compile-verified, matrix operand order checked by reading, not run on hardware
//
#include <hip/hip_runtime.h>

typedef __attribute__((ext_vector_type(16))) __bf16 v16bf;
typedef __attribute__((ext_vector_type(8)))  __bf16 v8bf;
typedef __attribute__((ext_vector_type(8)))  float  v8f;
typedef __attribute__((ext_vector_type(4)))  float  v4f;

#define B_ 16
#define N_ 1024
#define O_ 64
#define H_ 1024
#define D_ 256

// CDNA5 16x32 bf16 fragment: element e of a lane in half hv maps to
//   K = hv*8 + e          (e < 8)
//   K = 16 + hv*8 + (e-8) (e >= 8)
// => two contiguous 8-element (16-byte) runs per lane when operand is
//    K-contiguous.  Load them as b128s.

__device__ __forceinline__ v16bf frag_from_bf16(const __bf16* __restrict__ p, int hv) {
  v8bf a = *(const v8bf*)(p + hv * 8);
  v8bf b = *(const v8bf*)(p + 16 + hv * 8);
  v16bf f;
#pragma unroll
  for (int i = 0; i < 8; ++i) { f[i] = a[i]; f[8 + i] = b[i]; }
  return f;
}

__device__ __forceinline__ v16bf frag_from_f32(const float* __restrict__ p, int hv) {
  v4f a0 = *(const v4f*)(p + hv * 8);
  v4f a1 = *(const v4f*)(p + hv * 8 + 4);
  v4f b0 = *(const v4f*)(p + 16 + hv * 8);
  v4f b1 = *(const v4f*)(p + 16 + hv * 8 + 4);
  v16bf f;
#pragma unroll
  for (int i = 0; i < 4; ++i) {
    f[i]      = (__bf16)a0[i];
    f[4 + i]  = (__bf16)a1[i];
    f[8 + i]  = (__bf16)b0[i];
    f[12 + i] = (__bf16)b1[i];
  }
  return f;
}

__device__ __forceinline__ v8f wmma_bf16(const v16bf& a, const v16bf& b, const v8f& c) {
  return __builtin_amdgcn_wmma_f32_16x16x32_bf16(false, a, false, b, (short)0, c, false, false);
}

// ---------------------------------------------------------------------------
// Prep: W [H,D] f32  ->  Wt [D,H] bf16  (makes GEMM B-fragments K-contiguous)
// ---------------------------------------------------------------------------
__global__ __launch_bounds__(256) void wt_kernel(const float* __restrict__ W,
                                                 __bf16* __restrict__ Wt) {
  int i = blockIdx.x * 256 + threadIdx.x;  // over H*D, d fastest => coalesced read
  int d = i & (D_ - 1);
  int k = i >> 8;
  Wt[(long)d * H_ + k] = (__bf16)W[(long)k * D_ + d];
}

// ---------------------------------------------------------------------------
// Kernel 1: visual = l2norm(visual_tokens @ Wv + bv).
// Block = 16 rows x 256 cols; 8 waves x two 16x16 tiles.
// Writes bf16 visual in BOTH [B,N,D] and [B,D,N] layouts.
// ---------------------------------------------------------------------------
__global__ __launch_bounds__(256) void proj_visual_kernel(
    const float* __restrict__ X, const __bf16* __restrict__ WT,
    const float* __restrict__ bias, __bf16* __restrict__ visB,
    __bf16* __restrict__ visT) {
  __shared__ float tile[16 * 260];
  __shared__ float red[16 * 16];
  __shared__ float invn[16];

  const int m0   = blockIdx.x * 16;
  const int tid  = threadIdx.x;
  const int wave = tid >> 5;
  const int lane = tid & 31;
  const int hv   = lane >> 4;
  const int r16  = lane & 15;
  const int c0col = wave * 32 + r16;
  const int c1col = c0col + 16;

  v8f acc0, acc1;
  {
    float b0 = bias[c0col], b1 = bias[c1col];
#pragma unroll
    for (int v = 0; v < 8; ++v) { acc0[v] = b0; acc1[v] = b1; }
  }

  const float*  Xr  = X + (long)(m0 + r16) * H_;
  const __bf16* Wr0 = WT + (long)c0col * H_;
  const __bf16* Wr1 = WT + (long)c1col * H_;
  for (int k0 = 0; k0 < H_; k0 += 32) {
    __builtin_prefetch(Xr + k0 + 128, 0, 3);
    v16bf a  = frag_from_f32(Xr + k0, hv);
    v16bf b0 = frag_from_bf16(Wr0 + k0, hv);
    v16bf b1 = frag_from_bf16(Wr1 + k0, hv);
    acc0 = wmma_bf16(a, b0, acc0);
    acc1 = wmma_bf16(a, b1, acc1);
  }
#pragma unroll
  for (int v = 0; v < 8; ++v) {
    tile[(hv * 8 + v) * 260 + c0col] = acc0[v];
    tile[(hv * 8 + v) * 260 + c1col] = acc1[v];
  }
  __syncthreads();

  const int r = tid >> 4, j = tid & 15;
  {
    float s = 0.f;
    for (int c = j; c < D_; c += 16) { float x = tile[r * 260 + c]; s += x * x; }
    red[r * 16 + j] = s;
  }
  __syncthreads();
  if (tid < 16) {
    float s = 0.f;
#pragma unroll
    for (int q = 0; q < 16; ++q) s += red[tid * 16 + q];
    invn[tid] = 1.0f / fmaxf(sqrtf(s), 1e-12f);
  }
  __syncthreads();
  {
    // [B,N,D] layout: consecutive tids -> consecutive d
    const int m = m0 + r;
    float iv = invn[r];
    for (int c = j; c < D_; c += 16)
      visB[(long)m * D_ + c] = (__bf16)(tile[r * 260 + c] * iv);
  }
  {
    // [B,D,N] layout: consecutive tids -> consecutive n
    const int n2 = tid & 15, c2 = tid >> 4;
    const int bb = m0 >> 10;           // batch (N_=1024, 16-row tiles never straddle)
    const int nbase = (m0 & (N_ - 1)) + n2;
    float iv = invn[n2];
    for (int cc = c2; cc < D_; cc += 16)
      visT[((long)bb * D_ + cc) * N_ + nbase] = (__bf16)(tile[n2 * 260 + cc] * iv);
  }
}

// ---------------------------------------------------------------------------
// Kernel 2: phrase = l2norm(phrase_states @ Wp + bp)  (f32, for MSE target)
//           prototypes = l2norm(proto[o] + phrase)    (bf16, for attention A)
// ---------------------------------------------------------------------------
__global__ __launch_bounds__(256) void proj_phrase_kernel(
    const float* __restrict__ X, const __bf16* __restrict__ WT,
    const float* __restrict__ bias, const float* __restrict__ proto,
    float* __restrict__ phrase_out, __bf16* __restrict__ proto_out) {
  __shared__ float tile[16 * 260];
  __shared__ float red[16 * 16];
  __shared__ float invn[16];

  const int m0   = blockIdx.x * 16;
  const int tid  = threadIdx.x;
  const int wave = tid >> 5;
  const int lane = tid & 31;
  const int hv   = lane >> 4;
  const int r16  = lane & 15;
  const int c0col = wave * 32 + r16;
  const int c1col = c0col + 16;

  v8f acc0, acc1;
  {
    float b0 = bias[c0col], b1 = bias[c1col];
#pragma unroll
    for (int v = 0; v < 8; ++v) { acc0[v] = b0; acc1[v] = b1; }
  }
  const float*  Xr  = X + (long)(m0 + r16) * H_;
  const __bf16* Wr0 = WT + (long)c0col * H_;
  const __bf16* Wr1 = WT + (long)c1col * H_;
  for (int k0 = 0; k0 < H_; k0 += 32) {
    v16bf a  = frag_from_f32(Xr + k0, hv);
    v16bf b0 = frag_from_bf16(Wr0 + k0, hv);
    v16bf b1 = frag_from_bf16(Wr1 + k0, hv);
    acc0 = wmma_bf16(a, b0, acc0);
    acc1 = wmma_bf16(a, b1, acc1);
  }
#pragma unroll
  for (int v = 0; v < 8; ++v) {
    tile[(hv * 8 + v) * 260 + c0col] = acc0[v];
    tile[(hv * 8 + v) * 260 + c1col] = acc1[v];
  }
  __syncthreads();

  const int r = tid >> 4, j = tid & 15;
  {
    float s = 0.f;
    for (int c = j; c < D_; c += 16) { float x = tile[r * 260 + c]; s += x * x; }
    red[r * 16 + j] = s;
  }
  __syncthreads();
  if (tid < 16) {
    float s = 0.f;
#pragma unroll
    for (int q = 0; q < 16; ++q) s += red[tid * 16 + q];
    invn[tid] = 1.0f / fmaxf(sqrtf(s), 1e-12f);
  }
  __syncthreads();
  {
    const int m = m0 + r;
    const int o = m & (O_ - 1);
    float iv = invn[r];
    for (int c = j; c < D_; c += 16) {
      float ph = tile[r * 260 + c] * iv;
      phrase_out[(long)m * D_ + c] = ph;
      tile[r * 260 + c] = ph + proto[o * D_ + c];
    }
  }
  __syncthreads();
  {
    float s = 0.f;
    for (int c = j; c < D_; c += 16) { float x = tile[r * 260 + c]; s += x * x; }
    red[r * 16 + j] = s;
  }
  __syncthreads();
  if (tid < 16) {
    float s = 0.f;
#pragma unroll
    for (int q = 0; q < 16; ++q) s += red[tid * 16 + q];
    invn[tid] = 1.0f / fmaxf(sqrtf(s), 1e-12f);
  }
  __syncthreads();
  {
    const int m = m0 + r;
    float iv = invn[r];
    for (int c = j; c < D_; c += 16)
      proto_out[(long)m * D_ + c] = (__bf16)(tile[r * 260 + c] * iv);
  }
}

// ---------------------------------------------------------------------------
// Kernel 3: attention logits (prototypes . visual, K=D=256) + softmax over N
// + entropy.  Block = (b, o-tile) x 1024 N; logits live in WMMA registers.
// Writes f32 attention (output) and bf16 attention (evidence GEMM A).
// ---------------------------------------------------------------------------
__global__ __launch_bounds__(256) void attention_kernel(
    const __bf16* __restrict__ protos, const __bf16* __restrict__ visB,
    float* __restrict__ att_out, __bf16* __restrict__ attB,
    float* __restrict__ ent_acc) {
  __shared__ float part[256 * 8];
  __shared__ float rowval[16];
  __shared__ float red[256];

  const int b    = blockIdx.x >> 2;
  const int ot   = blockIdx.x & 3;
  const int tid  = threadIdx.x;
  const int wave = tid >> 5;
  const int lane = tid & 31;
  const int hv   = lane >> 4;
  const int r16  = lane & 15;
  const int mrow = b * O_ + ot * 16 + r16;

  v8f acc[8];
#pragma unroll
  for (int t = 0; t < 8; ++t) {
#pragma unroll
    for (int v = 0; v < 8; ++v) acc[t][v] = 0.f;
  }

  const __bf16* Ar = protos + (long)mrow * D_;
  for (int k0 = 0; k0 < D_; k0 += 32) {
    v16bf a = frag_from_bf16(Ar + k0, hv);
#pragma unroll
    for (int t = 0; t < 8; ++t) {
      const int col = wave * 128 + t * 16 + r16;  // token
      v16bf bm = frag_from_bf16(visB + ((long)b * N_ + col) * D_ + k0, hv);
      acc[t] = wmma_bf16(a, bm, acc[t]);
    }
  }

  // row max (row = hv*8+v, col = wave*128 + t*16 + r16)
#pragma unroll
  for (int v = 0; v < 8; ++v) {
    float m = -3.0e38f;
#pragma unroll
    for (int t = 0; t < 8; ++t) m = fmaxf(m, acc[t][v]);
    part[tid * 8 + v] = m;
  }
  __syncthreads();
  if (tid < 16) {
    const int rr = tid, rhv = rr >> 3, rv = rr & 7;
    float m = -3.0e38f;
    for (int w = 0; w < 8; ++w)
      for (int l = 0; l < 16; ++l)
        m = fmaxf(m, part[(w * 32 + rhv * 16 + l) * 8 + rv]);
    rowval[rr] = m;
  }
  __syncthreads();
  float rmax[8];
#pragma unroll
  for (int v = 0; v < 8; ++v) rmax[v] = rowval[hv * 8 + v];
  __syncthreads();

#pragma unroll
  for (int v = 0; v < 8; ++v) {
    float s = 0.f;
#pragma unroll
    for (int t = 0; t < 8; ++t) {
      float e = __expf(acc[t][v] - rmax[v]);
      acc[t][v] = e;
      s += e;
    }
    part[tid * 8 + v] = s;
  }
  __syncthreads();
  if (tid < 16) {
    const int rr = tid, rhv = rr >> 3, rv = rr & 7;
    float s = 0.f;
    for (int w = 0; w < 8; ++w)
      for (int l = 0; l < 16; ++l)
        s += part[(w * 32 + rhv * 16 + l) * 8 + rv];
    rowval[rr] = 1.0f / s;
  }
  __syncthreads();
  float rinv[8];
#pragma unroll
  for (int v = 0; v < 8; ++v) rinv[v] = rowval[hv * 8 + v];

  float ent = 0.f;
#pragma unroll
  for (int v = 0; v < 8; ++v) {
    const int orow = b * O_ + ot * 16 + hv * 8 + v;
#pragma unroll
    for (int t = 0; t < 8; ++t) {
      const int col = wave * 128 + t * 16 + r16;
      float p = acc[t][v] * rinv[v];
      att_out[(long)orow * N_ + col] = p;
      attB[(long)orow * N_ + col]    = (__bf16)p;
      ent += p * __logf(fmaxf(p, 1e-8f));
    }
  }
  red[tid] = ent;
  __syncthreads();
  if (tid == 0) {
    float s = 0.f;
    for (int q = 0; q < 256; ++q) s += red[q];
    atomicAdd(ent_acc, s);
  }
}

// ---------------------------------------------------------------------------
// Kernel 4: evidence = attention @ visual (K=N=1024, via visT [B,D,N])
//           + classifier logits + MSE.
// ---------------------------------------------------------------------------
__global__ __launch_bounds__(256) void evidence_kernel(
    const __bf16* __restrict__ attB, const __bf16* __restrict__ visT,
    const float* __restrict__ phrase, const float* __restrict__ labels,
    const float* __restrict__ Wc, const float* __restrict__ bc,
    float* __restrict__ logits_out, float* __restrict__ evid_out,
    float* __restrict__ mse_acc) {
  __shared__ float tile[16 * 260];
  __shared__ float red[16 * 16];

  const int b    = blockIdx.x >> 2;
  const int mt   = blockIdx.x & 3;
  const int tid  = threadIdx.x;
  const int wave = tid >> 5;
  const int lane = tid & 31;
  const int hv   = lane >> 4;
  const int r16  = lane & 15;
  const int c0col = wave * 32 + r16;
  const int c1col = c0col + 16;
  const int arow  = b * O_ + mt * 16 + r16;

  v8f acc0, acc1;
#pragma unroll
  for (int v = 0; v < 8; ++v) { acc0[v] = 0.f; acc1[v] = 0.f; }

  const __bf16* Ar  = attB + (long)arow * N_;
  const __bf16* Br0 = visT + ((long)b * D_ + c0col) * N_;
  const __bf16* Br1 = visT + ((long)b * D_ + c1col) * N_;
  for (int k0 = 0; k0 < N_; k0 += 32) {
    __builtin_prefetch(Ar + k0 + 128, 0, 3);
    v16bf a  = frag_from_bf16(Ar + k0, hv);
    v16bf b0 = frag_from_bf16(Br0 + k0, hv);
    v16bf b1 = frag_from_bf16(Br1 + k0, hv);
    acc0 = wmma_bf16(a, b0, acc0);
    acc1 = wmma_bf16(a, b1, acc1);
  }
#pragma unroll
  for (int v = 0; v < 8; ++v) {
    tile[(hv * 8 + v) * 260 + c0col] = acc0[v];
    tile[(hv * 8 + v) * 260 + c1col] = acc1[v];
  }
  __syncthreads();

  const int r = tid >> 4, j = tid & 15;
  const int m = b * O_ + mt * 16 + r;
  float dotp = 0.f, mse = 0.f;
  const float lab = labels[m];
  for (int c = j; c < D_; c += 16) {
    float e = tile[r * 260 + c];
    evid_out[(long)m * D_ + c] = e;
    dotp += e * Wc[c];
    float dd = e - lab * phrase[(long)m * D_ + c];
    mse += dd * dd;
  }
  red[r * 16 + j] = dotp;
  __syncthreads();
  if (j == 0) {
    float s = 0.f;
#pragma unroll
    for (int q = 0; q < 16; ++q) s += red[r * 16 + q];
    logits_out[m] = s + bc[0];
  }
  __syncthreads();
  red[tid] = mse;
  __syncthreads();
  if (tid == 0) {
    float s = 0.f;
    for (int q = 0; q < 256; ++q) s += red[q];
    atomicAdd(mse_acc, s);
  }
}

__global__ void init_acc_kernel(float* acc) {
  if (threadIdx.x < 4) acc[threadIdx.x] = 0.f;
}

__global__ void finalize_kernel(const float* __restrict__ acc,
                                float* __restrict__ loss_out) {
  if (threadIdx.x == 0) {
    float sparse = -acc[0] / (float)(B_ * O_);
    float ploss  =  acc[1] / (float)(B_ * O_ * D_);
    loss_out[0] = ploss + 0.01f * sparse;
  }
}

extern "C" void kernel_launch(void* const* d_in, const int* in_sizes, int n_in,
                              void* d_out, int out_size, void* d_ws, size_t ws_size,
                              hipStream_t stream) {
  const float* visual_tokens = (const float*)d_in[0];
  const float* phrase_states = (const float*)d_in[1];
  const float* labels        = (const float*)d_in[2];
  const float* Wv            = (const float*)d_in[3];
  const float* bv            = (const float*)d_in[4];
  const float* Wp            = (const float*)d_in[5];
  const float* bp            = (const float*)d_in[6];
  const float* proto         = (const float*)d_in[7];
  const float* Wc            = (const float*)d_in[8];
  const float* bc            = (const float*)d_in[9];

  float* out        = (float*)d_out;
  float* logits_out = out;                                 // [B,O]
  float* att_out    = out + (size_t)B_ * O_;               // [B,O,N]
  float* evid_out   = att_out + (size_t)B_ * O_ * N_;      // [B,O,D]
  float* loss_out   = evid_out + (size_t)B_ * O_ * D_;     // scalar

  // workspace layout: f32 first, then 16B-aligned bf16 arrays
  float* phrase = (float*)d_ws;                            // [B,O,D] f32
  float* acc    = phrase + (size_t)B_ * O_ * D_;           // 4 floats (padded)
  __bf16* visB  = (__bf16*)(acc + 4);                      // [B,N,D]
  __bf16* visT  = visB + (size_t)B_ * N_ * D_;             // [B,D,N]
  __bf16* protB = visT + (size_t)B_ * N_ * D_;             // [B,O,D]
  __bf16* attB  = protB + (size_t)B_ * O_ * D_;            // [B,O,N]
  __bf16* WvT   = attB + (size_t)B_ * O_ * N_;             // [D,H]
  __bf16* WpT   = WvT + (size_t)H_ * D_;                   // [D,H]

  init_acc_kernel<<<1, 32, 0, stream>>>(acc);
  wt_kernel<<<(H_ * D_) / 256, 256, 0, stream>>>(Wv, WvT);
  wt_kernel<<<(H_ * D_) / 256, 256, 0, stream>>>(Wp, WpT);
  proj_visual_kernel<<<(B_ * N_) / 16, 256, 0, stream>>>(visual_tokens, WvT, bv,
                                                         visB, visT);
  proj_phrase_kernel<<<(B_ * O_) / 16, 256, 0, stream>>>(phrase_states, WpT, bp, proto,
                                                         phrase, protB);
  attention_kernel<<<B_ * (O_ / 16), 256, 0, stream>>>(protB, visB, att_out, attB,
                                                       acc + 0);
  evidence_kernel<<<B_ * (O_ / 16), 256, 0, stream>>>(attB, visT, phrase, labels,
                                                      Wc, bc, logits_out, evid_out,
                                                      acc + 1);
  finalize_kernel<<<1, 32, 0, stream>>>(acc, loss_out);
}
